// GAT_7103875907620
// MI455X (gfx1250) — matrix operand order repaired and checked by reading, not compiled, not run on hardware
//
#include <hip/hip_runtime.h>
#include <hip/hip_bf16.h>

typedef __attribute__((ext_vector_type(16))) _Float16 v16h;
typedef __attribute__((ext_vector_type(8)))  _Float16 v8h;
typedef __attribute__((ext_vector_type(8)))  float    v8f;

union HV16 { v16h v; v8h h[2]; };

#define NEG_SLOPE 0.2f

// ---------------------------------------------------------------- utilities
__global__ void gat_fill(float* __restrict__ p, float v, long long n) {
    long long i = (long long)blockIdx.x * blockDim.x + threadIdx.x;
    if (i < n) p[i] = v;
}

__device__ __forceinline__ void atomicMaxF(float* addr, float val) {
    int* ai = (int*)addr;
    int old = *ai;
    while (__int_as_float(old) < val) {
        int assumed = old;
        old = atomicCAS(ai, assumed, __float_as_int(val));
        if (old == assumed) break;
    }
}

// ---------------------------------------------------------------- GEMM1: h1 = x[N,75] @ W1[75,64]   (WMMA f16 -> f32)
#define KP1 96
__global__ void __launch_bounds__(256) gat_gemm1(const float* __restrict__ x,
                                                 const float* __restrict__ W,
                                                 float* __restrict__ h1, int n) {
    __shared__ alignas(16) _Float16 sX[128][KP1];      // row-major rows of x (A operand)
    __shared__ alignas(16) _Float16 sWb[64][KP1];      // column-major W (B operand order)
    const int tid = threadIdx.x;
    const int rowBase0 = blockIdx.x * 128;

    for (int i = tid; i < 64 * KP1; i += 256) {
        int c = i / KP1, k = i % KP1;
        sWb[c][k] = (_Float16)((k < 75) ? W[k * 64 + c] : 0.0f);
    }
    for (int i = tid; i < 128 * KP1; i += 256) {
        int r = i / KP1, k = i % KP1;
        int row = rowBase0 + r;
        sX[r][k] = (_Float16)((row < n && k < 75) ? x[(long long)row * 75 + k] : 0.0f);
    }
    __syncthreads();

    const int lane = tid & 31;
    const int wave = tid >> 5;
    const int rloc = wave * 16;          // wave's 16-row slice of the 128-row slab
    const int mrow = lane & 15;          // M for A / N for B,C
    const int kgrp = lane >> 4;          // 0 or 1

    v8f zero = {};
    v8f acc[4];
    #pragma unroll
    for (int t = 0; t < 4; ++t) acc[t] = zero;

    #pragma unroll
    for (int s = 0; s < 3; ++s) {
        // A 16x32 f16 layout: v16h elems 0..7  <- K = 32s + kgrp*8 + 0..7   (contiguous)
        //                     elems 8..15 <- K = 32s + 16 + kgrp*8 + 0..7   (contiguous)
        HV16 a;
        a.h[0] = *(const v8h*)&sX[rloc + mrow][32 * s + kgrp * 8];
        a.h[1] = *(const v8h*)&sX[rloc + mrow][32 * s + 16 + kgrp * 8];
        #pragma unroll
        for (int t = 0; t < 4; ++t) {
            // B 32x16 f16 layout: elems 0..7 <- K = 32s + kgrp*8 + 0..7, elems 8..15 <- K = 32s + 16 + kgrp*8 + 0..7
            HV16 b;
            b.h[0] = *(const v8h*)&sWb[t * 16 + mrow][32 * s + kgrp * 8];
            b.h[1] = *(const v8h*)&sWb[t * 16 + mrow][32 * s + 16 + kgrp * 8];
            acc[t] = __builtin_amdgcn_wmma_f32_16x16x32_f16(
                false, a.v, false, b.v, (short)0, acc[t], false, false);
        }
    }
    // C/D layout: row = vgpr r + 8*kgrp, col = lane&15
    #pragma unroll
    for (int t = 0; t < 4; ++t)
        #pragma unroll
        for (int r = 0; r < 8; ++r) {
            int row = rowBase0 + rloc + r + 8 * kgrp;
            if (row < n) h1[(long long)row * 64 + t * 16 + mrow] = acc[t][r];
        }
}

// ---------------------------------------------------------------- GEMM2: h2 = x2[N,64] @ W2[64,128]
__global__ void __launch_bounds__(256) gat_gemm2(const _Float16* __restrict__ x2h,
                                                 const float* __restrict__ W,
                                                 float* __restrict__ h2, int n) {
    __shared__ alignas(16) _Float16 sX[128][64];
    __shared__ alignas(16) _Float16 sWb[128][64];      // column-major W2 (B operand order)
    const int tid = threadIdx.x;
    const int rowBase0 = blockIdx.x * 128;

    for (int i = tid; i < 128 * 64; i += 256) {
        int c = i / 64, k = i % 64;
        sWb[c][k] = (_Float16)W[k * 128 + c];
    }
    for (int i = tid; i < 128 * 64; i += 256) {
        int r = i / 64, k = i % 64;
        int row = rowBase0 + r;
        sX[r][k] = (row < n) ? x2h[(long long)row * 64 + k] : (_Float16)0.0f;
    }
    __syncthreads();

    const int lane = tid & 31;
    const int wave = tid >> 5;
    const int rloc = wave * 16;
    const int mrow = lane & 15;
    const int kgrp = lane >> 4;

    v8f zero = {};
    v8f acc[8];
    #pragma unroll
    for (int t = 0; t < 8; ++t) acc[t] = zero;

    #pragma unroll
    for (int s = 0; s < 2; ++s) {
        HV16 a;
        a.h[0] = *(const v8h*)&sX[rloc + mrow][32 * s + kgrp * 8];
        a.h[1] = *(const v8h*)&sX[rloc + mrow][32 * s + 16 + kgrp * 8];
        #pragma unroll
        for (int t = 0; t < 8; ++t) {
            HV16 b;
            b.h[0] = *(const v8h*)&sWb[t * 16 + mrow][32 * s + kgrp * 8];
            b.h[1] = *(const v8h*)&sWb[t * 16 + mrow][32 * s + 16 + kgrp * 8];
            acc[t] = __builtin_amdgcn_wmma_f32_16x16x32_f16(
                false, a.v, false, b.v, (short)0, acc[t], false, false);
        }
    }
    #pragma unroll
    for (int t = 0; t < 8; ++t)
        #pragma unroll
        for (int r = 0; r < 8; ++r) {
            int row = rowBase0 + rloc + r + 8 * kgrp;
            if (row < n) h2[(long long)row * 128 + t * 16 + mrow] = acc[t][r];
        }
}

// ---------------------------------------------------------------- attention scalars
__global__ void gat_att1(const float* __restrict__ h1, const float* __restrict__ atts,
                         const float* __restrict__ attd, float* __restrict__ as,
                         float* __restrict__ ad, int n) {
    long long idx = (long long)blockIdx.x * blockDim.x + threadIdx.x;
    if (idx >= (long long)n * 8) return;
    int node = (int)(idx >> 3), hd = (int)(idx & 7);
    float ss = 0.f, sd = 0.f;
    #pragma unroll
    for (int c = 0; c < 8; ++c) {
        float v = h1[(long long)node * 64 + hd * 8 + c];
        ss += v * atts[hd * 8 + c];
        sd += v * attd[hd * 8 + c];
    }
    as[idx] = ss; ad[idx] = sd;
}

__global__ void gat_att2(const float* __restrict__ h2, const float* __restrict__ atts,
                         const float* __restrict__ attd, float* __restrict__ as,
                         float* __restrict__ ad, int n) {
    int node = blockIdx.x * (blockDim.x >> 5) + (threadIdx.x >> 5);
    int lane = threadIdx.x & 31;
    if (node >= n) return;
    float ss = 0.f, sd = 0.f;
    for (int c = lane; c < 128; c += 32) {
        float v = h2[(long long)node * 128 + c];
        ss += v * atts[c];
        sd += v * attd[c];
    }
    #pragma unroll
    for (int off = 16; off > 0; off >>= 1) {
        ss += __shfl_down(ss, off);
        sd += __shfl_down(sd, off);
    }
    if (lane == 0) { as[node] = ss; ad[node] = sd; }
}

// ---------------------------------------------------------------- conv1 edge passes (H=8)
__device__ __forceinline__ void edge_sd(const int* ei, int nE, long long i, int& s, int& d) {
    if (i < nE) { s = ei[i]; d = ei[(long long)nE + i]; }
    else        { s = (int)(i - nE); d = s; }          // self loop
}

__global__ void gat_edge1_max(const int* __restrict__ ei, int nE, long long nTot,
                              const float* __restrict__ as, const float* __restrict__ ad,
                              float* __restrict__ emax) {
    long long i = (long long)blockIdx.x * blockDim.x + threadIdx.x;
    if (i >= nTot) return;
    int s, d; edge_sd(ei, nE, i, s, d);
    #pragma unroll
    for (int h = 0; h < 8; ++h) {
        float e = as[s * 8 + h] + ad[d * 8 + h];
        e = (e < 0.f) ? NEG_SLOPE * e : e;
        atomicMaxF(&emax[d * 8 + h], e);
    }
}

__global__ void gat_edge1_sum(const int* __restrict__ ei, int nE, long long nTot,
                              const float* __restrict__ as, const float* __restrict__ ad,
                              const float* __restrict__ emax, float* __restrict__ den) {
    long long i = (long long)blockIdx.x * blockDim.x + threadIdx.x;
    if (i >= nTot) return;
    int s, d; edge_sd(ei, nE, i, s, d);
    #pragma unroll
    for (int h = 0; h < 8; ++h) {
        float e = as[s * 8 + h] + ad[d * 8 + h];
        e = (e < 0.f) ? NEG_SLOPE * e : e;
        atomicAdd(&den[d * 8 + h], __expf(e - emax[d * 8 + h]));
    }
}

__global__ void gat_edge1_msg(const int* __restrict__ ei, int nE, long long nTot,
                              const float* __restrict__ as, const float* __restrict__ ad,
                              const float* __restrict__ emax, const float* __restrict__ den,
                              const float* __restrict__ h1, float* __restrict__ out1) {
    long long idx = (long long)blockIdx.x * blockDim.x + threadIdx.x;
    if (idx >= nTot * 64) return;
    long long i = idx >> 6;
    int c = (int)(idx & 63), h = c >> 3;
    int s, d; edge_sd(ei, nE, i, s, d);
    float e = as[s * 8 + h] + ad[d * 8 + h];
    e = (e < 0.f) ? NEG_SLOPE * e : e;
    float alpha = __expf(e - emax[d * 8 + h]) / den[d * 8 + h];
    atomicAdd(&out1[(long long)d * 64 + c], h1[(long long)s * 64 + c] * alpha);
}

// ---------------------------------------------------------------- x2 = ELU(out1 + bias1), stored f16
__global__ void gat_x2(const float* __restrict__ out1, const float* __restrict__ bias,
                       _Float16* __restrict__ x2h, int n) {
    long long idx = (long long)blockIdx.x * blockDim.x + threadIdx.x;
    if (idx >= (long long)n * 64) return;
    int c = (int)(idx & 63);
    float v = out1[idx] + bias[c];
    v = (v > 0.f) ? v : (__expf(v) - 1.0f);
    x2h[idx] = (_Float16)v;
}

// ---------------------------------------------------------------- conv2 edge passes (H=1, C=128)
__global__ void gat_edge2_max(const int* __restrict__ ei, int nE, long long nTot,
                              const float* __restrict__ as, const float* __restrict__ ad,
                              float* __restrict__ emax) {
    long long i = (long long)blockIdx.x * blockDim.x + threadIdx.x;
    if (i >= nTot) return;
    int s, d; edge_sd(ei, nE, i, s, d);
    float e = as[s] + ad[d];
    e = (e < 0.f) ? NEG_SLOPE * e : e;
    atomicMaxF(&emax[d], e);
}

__global__ void gat_edge2_sum(const int* __restrict__ ei, int nE, long long nTot,
                              const float* __restrict__ as, const float* __restrict__ ad,
                              const float* __restrict__ emax, float* __restrict__ den) {
    long long i = (long long)blockIdx.x * blockDim.x + threadIdx.x;
    if (i >= nTot) return;
    int s, d; edge_sd(ei, nE, i, s, d);
    float e = as[s] + ad[d];
    e = (e < 0.f) ? NEG_SLOPE * e : e;
    atomicAdd(&den[d], __expf(e - emax[d]));
}

__global__ void gat_edge2_msg(const int* __restrict__ ei, int nE, long long nTot,
                              const float* __restrict__ as, const float* __restrict__ ad,
                              const float* __restrict__ emax, const float* __restrict__ den,
                              const float* __restrict__ h2, float* __restrict__ out2) {
    long long idx = (long long)blockIdx.x * blockDim.x + threadIdx.x;
    if (idx >= nTot * 128) return;
    long long i = idx >> 7;
    int c = (int)(idx & 127);
    int s, d; edge_sd(ei, nE, i, s, d);
    float e = as[s] + ad[d];
    e = (e < 0.f) ? NEG_SLOPE * e : e;
    float alpha = __expf(e - emax[d]) / den[d];
    atomicAdd(&out2[(long long)d * 128 + c], h2[(long long)s * 128 + c] * alpha);
}

// ---------------------------------------------------------------- pooling (fused bias2) + head
__global__ void gat_pool(const float* __restrict__ out2, const float* __restrict__ bias,
                         const int* __restrict__ batch, float* __restrict__ pooled, int n) {
    long long idx = (long long)blockIdx.x * blockDim.x + threadIdx.x;
    if (idx >= (long long)n * 128) return;
    int node = (int)(idx >> 7), c = (int)(idx & 127);
    atomicAdd(&pooled[(long long)batch[node] * 128 + c], out2[idx] + bias[c]);
}

__global__ void gat_final(const float* __restrict__ pooled, const float* __restrict__ Wg,
                          const float* __restrict__ bg, float* __restrict__ y, int G) {
    int g = blockIdx.x;
    if (g >= G) return;
    int tid = threadIdx.x, lane = tid & 31, wave = tid >> 5;
    float v = pooled[(long long)g * 128 + tid] * Wg[tid];
    #pragma unroll
    for (int off = 16; off > 0; off >>= 1) v += __shfl_down(v, off);
    __shared__ float red[4];
    if (lane == 0) red[wave] = v;
    __syncthreads();
    if (tid == 0) y[g] = red[0] + red[1] + red[2] + red[3] + bg[0];
}

// ---------------------------------------------------------------- launcher
extern "C" void kernel_launch(void* const* d_in, const int* in_sizes, int n_in,
                              void* d_out, int out_size, void* d_ws, size_t ws_size,
                              hipStream_t stream) {
    const float* x     = (const float*)d_in[0];
    const int*   ei    = (const int*)d_in[1];
    const int*   batch = (const int*)d_in[2];
    const float* W1    = (const float*)d_in[3];
    const float* atts1 = (const float*)d_in[4];
    const float* attd1 = (const float*)d_in[5];
    const float* b1    = (const float*)d_in[6];
    const float* W2    = (const float*)d_in[7];
    const float* atts2 = (const float*)d_in[8];
    const float* attd2 = (const float*)d_in[9];
    const float* b2    = (const float*)d_in[10];
    const float* Wg    = (const float*)d_in[11];
    const float* bg    = (const float*)d_in[12];
    float* y = (float*)d_out;

    const int N = in_sizes[0] / 75;
    const int E = in_sizes[1] / 2;
    const long long ET = (long long)E + N;     // + self loops
    const int G = out_size;
    const size_t nn = (size_t)N;

    // workspace layout (floats)
    float* base  = (float*)d_ws;
    float* h1    = base;                 // 64N
    float* as1   = h1    + 64 * nn;      // 8N
    float* ad1   = as1   + 8 * nn;       // 8N
    float* emax1 = ad1   + 8 * nn;       // 8N
    float* den1  = emax1 + 8 * nn;       // 8N
    float* out1  = den1  + 8 * nn;       // 64N
    _Float16* x2h = (_Float16*)(out1 + 64 * nn);   // 64N halves (32N floats)
    float* h2    = out1 + 64 * nn + 32 * nn;       // 128N
    float* as2   = h2    + 128 * nn;     // N
    float* ad2   = as2   + nn;           // N
    float* emax2 = ad2   + nn;           // N
    float* den2  = emax2 + nn;           // N
    float* out2  = den2  + nn;           // 128N
    float* pooled = out2 + 128 * nn;     // 128G

    auto cdiv = [](long long a, long long b) -> unsigned { return (unsigned)((a + b - 1) / b); };

    // init accumulators
    gat_fill<<<cdiv(8 * (long long)nn, 256), 256, 0, stream>>>(emax1, -3.0e38f, 8 * (long long)nn);
    gat_fill<<<cdiv(8 * (long long)nn, 256), 256, 0, stream>>>(den1, 0.0f, 8 * (long long)nn);
    gat_fill<<<cdiv(64 * (long long)nn, 256), 256, 0, stream>>>(out1, 0.0f, 64 * (long long)nn);
    gat_fill<<<cdiv((long long)nn, 256), 256, 0, stream>>>(emax2, -3.0e38f, (long long)nn);
    gat_fill<<<cdiv((long long)nn, 256), 256, 0, stream>>>(den2, 0.0f, (long long)nn);
    gat_fill<<<cdiv(128 * (long long)nn, 256), 256, 0, stream>>>(out2, 0.0f, 128 * (long long)nn);
    gat_fill<<<cdiv(128LL * G, 256), 256, 0, stream>>>(pooled, 0.0f, 128LL * G);

    // conv1
    gat_gemm1<<<cdiv(N, 128), 256, 0, stream>>>(x, W1, h1, N);
    gat_att1<<<cdiv(8 * (long long)nn, 256), 256, 0, stream>>>(h1, atts1, attd1, as1, ad1, N);
    gat_edge1_max<<<cdiv(ET, 256), 256, 0, stream>>>(ei, E, ET, as1, ad1, emax1);
    gat_edge1_sum<<<cdiv(ET, 256), 256, 0, stream>>>(ei, E, ET, as1, ad1, emax1, den1);
    gat_edge1_msg<<<cdiv(ET * 64, 256), 256, 0, stream>>>(ei, E, ET, as1, ad1, emax1, den1, h1, out1);
    gat_x2<<<cdiv(64 * (long long)nn, 256), 256, 0, stream>>>(out1, b1, x2h, N);

    // conv2
    gat_gemm2<<<cdiv(N, 128), 256, 0, stream>>>(x2h, W2, h2, N);
    gat_att2<<<cdiv(N, 4), 128, 0, stream>>>(h2, atts2, attd2, as2, ad2, N);
    gat_edge2_max<<<cdiv(ET, 256), 256, 0, stream>>>(ei, E, ET, as2, ad2, emax2);
    gat_edge2_sum<<<cdiv(ET, 256), 256, 0, stream>>>(ei, E, ET, as2, ad2, emax2, den2);
    gat_edge2_msg<<<cdiv(ET * 128, 256), 256, 0, stream>>>(ei, E, ET, as2, ad2, emax2, den2, h2, out2);

    // pool + head
    gat_pool<<<cdiv(128 * (long long)nn, 256), 256, 0, stream>>>(out2, b2, batch, pooled, N);
    gat_final<<<G, 128, 0, stream>>>(pooled, Wg, bg, y, G);
}
